// Generator_47072841565036
// MI455X (gfx1250) — matrix operand order repaired and checked by reading, not compiled, not run on hardware
//
#include <hip/hip_runtime.h>
#include <hip/hip_bf16.h>
#include <math.h>

typedef __attribute__((ext_vector_type(8)))  _Float16 v8h;
typedef __attribute__((ext_vector_type(16))) _Float16 v16h;
typedef __attribute__((ext_vector_type(8)))  float    v8f;

#define BATCH 256
#define NPIX  65536          // 256*256
#define NTOT  16777216       // 256 * 65536

// ---------------------------------------------------------------------------
// Kernel 1: build X0 = concat(embed[labels], noise) padded to 128 cols
// ---------------------------------------------------------------------------
__global__ void build_x0(const float* __restrict__ noise,
                         const int*   __restrict__ labels,
                         const float* __restrict__ embed,
                         float* __restrict__ X0)
{
    int row = blockIdx.x;          // 0..255
    int col = threadIdx.x;         // 0..127
    float v = 0.0f;
    if (col < 10)       v = embed[labels[row] * 10 + col];
    else if (col < 110) v = noise[row * 100 + (col - 10)];
    X0[row * 128 + col] = v;
}

// ---------------------------------------------------------------------------
// Kernel 2: fused Linear + BatchNorm1d(train stats, eps=0.8) + LeakyReLU(0.2)
// One workgroup owns 32 full columns (all 256 batch rows) -> batch stats local.
// ---------------------------------------------------------------------------
template<int K, int XS, int N>
__global__ void mlp_block(const float* __restrict__ X,
                          const float* __restrict__ W,
                          const float* __restrict__ b,
                          const float* __restrict__ g,
                          const float* __restrict__ be,
                          float* __restrict__ Y)
{
    __shared__ float h[BATCH * 32];          // 32 KB
    __shared__ float mu_s[32], rs_s[32];

    const int c   = threadIdx.x & 31;
    const int r0  = (threadIdx.x >> 5) * 32; // 8 row-groups of 32
    const int col = blockIdx.x * 32 + c;

    const float bias = b[col];
    for (int r = r0; r < r0 + 32; ++r) {
        float acc = bias;
        for (int k = 0; k < K; ++k)
            acc += X[r * XS + k] * W[k * N + col];
        h[r * 32 + c] = acc;
    }
    __syncthreads();

    if (threadIdx.x < 32) {
        float s = 0.0f;
        for (int r = 0; r < BATCH; ++r) s += h[r * 32 + threadIdx.x];
        float mu = s * (1.0f / BATCH);
        float v = 0.0f;
        for (int r = 0; r < BATCH; ++r) {
            float d = h[r * 32 + threadIdx.x] - mu;
            v += d * d;
        }
        mu_s[threadIdx.x] = mu;
        rs_s[threadIdx.x] = rsqrtf(v * (1.0f / BATCH) + 0.8f);
    }
    __syncthreads();

    const float mu = mu_s[c], rs = rs_s[c], gg = g[col], bb = be[col];
    for (int r = r0; r < r0 + 32; ++r) {
        float hn = (h[r * 32 + c] - mu) * rs * gg + bb;
        Y[r * N + col] = (hn >= 0.0f) ? hn : 0.2f * hn;
    }
}

// ---------------------------------------------------------------------------
// Kernel 3: convert X4 (256x1024 fp32) -> f16 for WMMA A fragments
// ---------------------------------------------------------------------------
__global__ void cvt_f16(const float* __restrict__ X, _Float16* __restrict__ Xh)
{
    int i = blockIdx.x * blockDim.x + threadIdx.x;   // 0..262143
    Xh[i] = (_Float16)X[i];
}

// ---------------------------------------------------------------------------
// Kernel 4: img = tanh(X4 @ W5 + b5) via v_wmma_f32_16x16x32_f16.
// Each wave owns a 16-col strip; accumulates ALL 16 M-tiles per B fragment
// so the 268 MB W5 stream happens exactly once (HBM-bound at ~11.5 us).
// W5 is converted fp32->f16 in registers (no extra bandwidth).
// ---------------------------------------------------------------------------
__global__ void __launch_bounds__(256)
gemm5_wmma(const _Float16* __restrict__ Xh,
           const float* __restrict__ W5,
           const float* __restrict__ b5,
           float* __restrict__ img)
{
    const int lane = threadIdx.x & 31;
    const int wave = threadIdx.x >> 5;
    const int n0   = (blockIdx.x * 8 + wave) * 16;   // 512 blocks * 8 waves * 16
    const int col  = n0 + (lane & 15);
    const int kbA  = (lane & 16) ? 16 : 0;           // 16-bit A 16x32 layout (contiguous K)
    const int kbB  = (lane & 16) ? 16 : 0;           // 16-bit B 32x16 layout

    v8f c[16];
#pragma unroll
    for (int m = 0; m < 16; ++m) c[m] = (v8f){0,0,0,0,0,0,0,0};

    for (int kt = 0; kt < 1024; kt += 32) {
        // B fragment: lane holds column `col`, K = kt+kbB .. +15 (1 row/half)
        v16h bf;
#pragma unroll
        for (int h = 0; h < 16; ++h)
            bf[h] = (_Float16)W5[(size_t)(kt + kbB + h) * NPIX + col];

#pragma unroll
        for (int m = 0; m < 16; ++m) {
            const _Float16* aptr =
                Xh + (size_t)(m * 16 + (lane & 15)) * 1024 + kt + kbA;
            v8h lo = *reinterpret_cast<const v8h*>(aptr);        // K kbA+0..7
            v8h hi = *reinterpret_cast<const v8h*>(aptr + 8);    // K kbA+8..15
            v16h af = __builtin_shufflevector(lo, hi,
                        0,1,2,3,4,5,6,7,8,9,10,11,12,13,14,15);
            c[m] = __builtin_amdgcn_wmma_f32_16x16x32_f16(
                       false, af, false, bf, (short)0, c[m], false, false);
        }
    }

    const float bias = b5[col];
#pragma unroll
    for (int m = 0; m < 16; ++m) {
        const int rbase = m * 16 + ((lane & 16) ? 8 : 0);
#pragma unroll
        for (int i = 0; i < 8; ++i) {
            float v = tanhf(c[m][i] + bias);
            img[(size_t)(rbase + i) * NPIX + col] = v;
        }
    }
}

// ---------------------------------------------------------------------------
// Sorting machinery for _ste_gray: stable argsort(img) as u64 (key<<32|idx),
// plain f32 sort of random_images. Bitonic: LDS chunks + global cross passes.
// ---------------------------------------------------------------------------
__device__ __forceinline__ unsigned sortable_key(float f)
{
    unsigned u = __float_as_uint(f);
    return (u & 0x80000000u) ? ~u : (u | 0x80000000u);
}

__global__ void init_sort_bufs(const float* __restrict__ img,
                               const float* __restrict__ rnd,
                               unsigned long long* __restrict__ pairs,
                               float* __restrict__ pvals)
{
    size_t i = (size_t)blockIdx.x * blockDim.x + threadIdx.x;
    unsigned k = sortable_key(img[i]);
    pairs[i] = ((unsigned long long)k << 32) | (unsigned)(i & 0xFFFFu);
    pvals[i] = rnd[i];
}

template<typename T, int CH>
__global__ void bitonic_lds(T* __restrict__ data, int kl_start, int kl_end)
{
    __shared__ T sh[CH];                       // 32 KB for both instantiations
    const size_t base  = (size_t)blockIdx.x * CH;   // chunks never straddle rows
    const int    gbase = (int)(base & (NPIX - 1)); // in-row offset for direction

    for (int i = threadIdx.x; i < CH; i += blockDim.x) sh[i] = data[base + i];
    __syncthreads();

    for (int kl = kl_start; kl <= kl_end; ++kl) {
        const int k = 1 << kl;
        int j = k >> 1;
        if (j > (CH >> 1)) j = CH >> 1;
        for (; j > 0; j >>= 1) {
            for (int t = threadIdx.x; t < CH / 2; t += blockDim.x) {
                int i = ((t & ~(j - 1)) << 1) | (t & (j - 1));
                int l = i | j;
                bool up = (((gbase + i) & k) == 0);
                T a = sh[i], b = sh[l];
                if ((a > b) == up) { sh[i] = b; sh[l] = a; }
            }
            __syncthreads();
        }
    }
    for (int i = threadIdx.x; i < CH; i += blockDim.x) data[base + i] = sh[i];
}

template<typename T>
__global__ void bitonic_global(T* __restrict__ data, int k, int j)
{
    int t = blockIdx.x * blockDim.x + threadIdx.x;        // 0..32767 per row
    int i = ((t & ~(j - 1)) << 1) | (t & (j - 1));
    int l = i | j;
    size_t base = (size_t)blockIdx.y << 16;
    T a = data[base + i], b = data[base + l];
    bool up = ((i & k) == 0);
    if ((a > b) == up) { data[base + i] = b; data[base + l] = a; }
}

// out[row, argsort(g)[i]] = sort(p)[i]
__global__ void scatter_out(const unsigned long long* __restrict__ pairs,
                            const float* __restrict__ pvals,
                            float* __restrict__ outq)
{
    size_t i = (size_t)blockIdx.x * blockDim.x + threadIdx.x;
    unsigned idx = (unsigned)pairs[i] & 0xFFFFu;
    outq[(i & ~(size_t)0xFFFF) | idx] = pvals[i];
}

// ---------------------------------------------------------------------------
extern "C" void kernel_launch(void* const* d_in, const int* in_sizes, int n_in,
                              void* d_out, int out_size, void* d_ws, size_t ws_size,
                              hipStream_t stream)
{
    (void)in_sizes; (void)n_in; (void)out_size; (void)ws_size;

    const float* noise  = (const float*)d_in[0];
    const float* rimgs  = (const float*)d_in[1];
    const int*   labels = (const int*)  d_in[2];
    const float* embed  = (const float*)d_in[3];
    const float* W1 = (const float*)d_in[4];  const float* b1 = (const float*)d_in[5];
    const float* g1 = (const float*)d_in[6];  const float* be1= (const float*)d_in[7];
    const float* W2 = (const float*)d_in[8];  const float* b2 = (const float*)d_in[9];
    const float* g2 = (const float*)d_in[10]; const float* be2= (const float*)d_in[11];
    const float* W3 = (const float*)d_in[12]; const float* b3 = (const float*)d_in[13];
    const float* g3 = (const float*)d_in[14]; const float* be3= (const float*)d_in[15];
    const float* W4 = (const float*)d_in[16]; const float* b4 = (const float*)d_in[17];
    const float* g4 = (const float*)d_in[18]; const float* be4= (const float*)d_in[19];
    const float* W5 = (const float*)d_in[20]; const float* b5 = (const float*)d_in[21];

    float* out_q = (float*)d_out;            // img_quantized
    float* img   = (float*)d_out + NTOT;     // img

    char* ws = (char*)d_ws;
    float*    X0  = (float*)(ws);                         // 256*128
    float*    X1  = (float*)(ws + 131072);                // 256*128
    float*    X2  = (float*)(ws + 262144);                // 256*256
    float*    X3  = (float*)(ws + 524288);                // 256*512
    float*    X4  = (float*)(ws + 1048576);               // 256*1024
    _Float16* X4h = (_Float16*)(ws + 2097152);            // 256*1024 f16
    unsigned long long* pairs = (unsigned long long*)(ws + 2621440);   // 134 MB
    float*    pvals = (float*)(ws + 2621440 + (size_t)NTOT * 8);       // 67 MB

    // MLP trunk
    build_x0<<<BATCH, 128, 0, stream>>>(noise, labels, embed, X0);
    mlp_block<110, 128, 128> <<<128/32,  256, 0, stream>>>(X0, W1, b1, g1, be1, X1);
    mlp_block<128, 128, 256> <<<256/32,  256, 0, stream>>>(X1, W2, b2, g2, be2, X2);
    mlp_block<256, 256, 512> <<<512/32,  256, 0, stream>>>(X2, W3, b3, g3, be3, X3);
    mlp_block<512, 512, 1024><<<1024/32, 256, 0, stream>>>(X3, W4, b4, g4, be4, X4);
    cvt_f16<<<1024, 256, 0, stream>>>(X4, X4h);

    // Big WMMA GEMM + tanh -> img
    gemm5_wmma<<<512, 256, 0, stream>>>(X4h, W5, b5, img);

    // Sort buffers
    init_sort_bufs<<<NTOT / 256, 256, 0, stream>>>(img, rimgs, pairs, pvals);

    // argsort(img): u64 bitonic, CH=4096 pairs (32 KB LDS), 4096 chunks
    {
        const int CH = 4096;
        bitonic_lds<unsigned long long, 4096><<<NTOT / CH, 256, 0, stream>>>(pairs, 1, 12);
        for (int kl = 13; kl <= 16; ++kl) {
            for (int jl = kl - 1; jl >= 12; --jl)
                bitonic_global<unsigned long long>
                    <<<dim3(128, BATCH), 256, 0, stream>>>(pairs, 1 << kl, 1 << jl);
            bitonic_lds<unsigned long long, 4096><<<NTOT / CH, 256, 0, stream>>>(pairs, kl, kl);
        }
    }

    // sort(random_images): f32 bitonic, CH=8192 (32 KB LDS), 2048 chunks
    {
        const int CH = 8192;
        bitonic_lds<float, 8192><<<NTOT / CH, 256, 0, stream>>>(pvals, 1, 13);
        for (int kl = 14; kl <= 16; ++kl) {
            for (int jl = kl - 1; jl >= 13; --jl)
                bitonic_global<float>
                    <<<dim3(128, BATCH), 256, 0, stream>>>(pvals, 1 << kl, 1 << jl);
            bitonic_lds<float, 8192><<<NTOT / CH, 256, 0, stream>>>(pvals, kl, kl);
        }
    }

    // img_quantized
    scatter_out<<<NTOT / 256, 256, 0, stream>>>(pairs, pvals, out_q);
}
